// MultiHeadAttention_72069551227064
// MI455X (gfx1250) — matrix operand order repaired and checked by reading, not compiled, not run on hardware
//
#include <hip/hip_runtime.h>

// ---------------------------------------------------------------------------
// MultiHeadAttention (B=4, S=4096, D=256) for gfx1250 (MI455X).
// f32 -> bf16 WMMA (v_wmma_f32_16x16x32_bf16) with f32 accumulation,
// flash-attention fusion; K/V tiles staged with double-buffered
// GLOBAL_LOAD_ASYNC_TO_LDS_B128 (ASYNCcnt) when the builtin is available.
// ---------------------------------------------------------------------------

typedef __attribute__((ext_vector_type(8)))  __bf16 bf16x8;
typedef __attribute__((ext_vector_type(16))) __bf16 bf16x16;
typedef __attribute__((ext_vector_type(8)))  float  f32x8;

#define WMMA_BF16(a, b, c) \
    __builtin_amdgcn_wmma_f32_16x16x32_bf16(false, (a), false, (b), (short)0, (c), false, false)

#if __has_builtin(__builtin_amdgcn_global_load_async_to_lds_b128) && \
    __has_builtin(__builtin_amdgcn_s_wait_asynccnt)
#define HAVE_ASYNC_LDS 1
#else
#define HAVE_ASYNC_LDS 0
#endif

// Builtin signature (from clang diagnostic): non-const int4* in AS(1) and AS(3).
typedef __attribute__((__vector_size__(4 * sizeof(int)))) int v4i;
typedef __attribute__((address_space(1))) v4i g_v4i;
typedef __attribute__((address_space(3))) v4i l_v4i;
// Integer round-trip avoids addrspace-cast restrictions; generic LDS addresses
// truncate to the 32-bit LDS offset.
#define AS1P(p) ((g_v4i*)(unsigned long long)(p))
#define AS3P(p) ((l_v4i*)(unsigned int)(unsigned long long)(p))

union BfPack { bf16x16 v16; bf16x8 v8[2]; };

static __device__ __forceinline__ bf16x16 joinv(bf16x8 lo, bf16x8 hi) {
    BfPack u; u.v8[0] = lo; u.v8[1] = hi; return u.v16;
}

static __device__ __forceinline__ float rmax16(float v) {
    v = fmaxf(v, __shfl_xor(v, 1, 32));
    v = fmaxf(v, __shfl_xor(v, 2, 32));
    v = fmaxf(v, __shfl_xor(v, 4, 32));
    v = fmaxf(v, __shfl_xor(v, 8, 32));
    return v;
}
static __device__ __forceinline__ float rsum16(float v) {
    v += __shfl_xor(v, 1, 32);
    v += __shfl_xor(v, 2, 32);
    v += __shfl_xor(v, 4, 32);
    v += __shfl_xor(v, 8, 32);
    return v;
}

// Convert 16 consecutive f32 (given as 4 float4) into a bf16x16 fragment.
static __device__ __forceinline__ bf16x16 cvt16(float4 a, float4 b, float4 c, float4 d) {
    bf16x16 r;
    r[0]  = (__bf16)a.x; r[1]  = (__bf16)a.y; r[2]  = (__bf16)a.z; r[3]  = (__bf16)a.w;
    r[4]  = (__bf16)b.x; r[5]  = (__bf16)b.y; r[6]  = (__bf16)b.z; r[7]  = (__bf16)b.w;
    r[8]  = (__bf16)c.x; r[9]  = (__bf16)c.y; r[10] = (__bf16)c.z; r[11] = (__bf16)c.w;
    r[12] = (__bf16)d.x; r[13] = (__bf16)d.y; r[14] = (__bf16)d.z; r[15] = (__bf16)d.w;
    return r;
}

constexpr int Bn = 4;
constexpr int Sn = 4096;
constexpr int Dn = 256;

// ---------------------------------------------------------------------------
// Kernel 1: qkv = x @ W_qkv^T + b_qkv.
// Q (pre-scaled by 1/16) and K stored row-major bf16; V stored TRANSPOSED
// (VT[b][d][s], bf16) so the flash kernel stages it with contiguous b128 copies.
// ---------------------------------------------------------------------------
__global__ __launch_bounds__(256) void qkv_proj_kernel(
    const float* __restrict__ x, const float* __restrict__ Wqkv,
    const float* __restrict__ bqkv,
    __bf16* __restrict__ Q, __bf16* __restrict__ K, __bf16* __restrict__ VT)
{
    const int wave = threadIdx.x >> 5;
    const int lane = threadIdx.x & 31;
    const int hsel = lane >> 4;     // which 16-lane half
    const int lm   = lane & 15;

    const int rowTile = blockIdx.x / 6;              // 16384/16 = 1024 row tiles
    const int colTile = (blockIdx.x % 6) * 8 + wave; // 768/16 = 48 col tiles
    const int row0 = rowTile * 16;
    const int col0 = colTile * 16;

    const float* arow = x    + (size_t)(row0 + lm) * Dn;
    const float* brow = Wqkv + (size_t)(col0 + lm) * Dn;

    f32x8 acc = {};
#pragma unroll
    for (int f = 0; f < 8; ++f) {
        // A fragment: row (row0+lm); K = f*32 + 16*hsel + e (16 consecutive)
        const float4* ap0 = (const float4*)(arow + f * 32 + 16 * hsel);
        const float4* ap1 = (const float4*)(arow + f * 32 + 16 * hsel + 8);
        bf16x16 afrag = cvt16(ap0[0], ap0[1], ap1[0], ap1[1]);
        // B fragment: column (col0+lm); K = f*32 + 16*hsel + e (16 consecutive)
        const float4* bp = (const float4*)(brow + f * 32 + 16 * hsel);
        bf16x16 bfrag = cvt16(bp[0], bp[1], bp[2], bp[3]);
        acc = WMMA_BF16(afrag, bfrag, acc);
    }

    const float bias = bqkv[col0 + lm];
    if (col0 < 512) {
        __bf16* dst = Q;
        int cc = col0 + lm;
        float sc = 0.0625f;                       // fold 1/sqrt(256) into Q
        if (col0 >= 256) { dst = K; cc -= 256; sc = 1.0f; }
#pragma unroll
        for (int r = 0; r < 8; ++r) {
            const int row = row0 + 8 * hsel + r;
            dst[(size_t)row * Dn + cc] = (__bf16)((acc[r] + bias) * sc);
        }
    } else {
        // V, stored transposed: VT[b][d][s]; 8 consecutive s per lane -> b128 store
        const int cc   = col0 + lm - 512;
        const int bidx = row0 >> 12;              // token row / 4096
        const int s0   = (row0 & 4095) + 8 * hsel;
        bf16x8 pk;
#pragma unroll
        for (int r = 0; r < 8; ++r) pk[r] = (__bf16)(acc[r] + bias);
        *(bf16x8*)(VT + (size_t)bidx * Dn * Sn + (size_t)cc * Sn + s0) = pk;
    }
}

// ---------------------------------------------------------------------------
// Kernel 2: flash attention. One block = one batch x 128-query block.
// 8 waves x 16 query rows; double-buffered 32-key blocks staged in LDS
// via async-to-LDS copies (fallback: synchronous vector copies).
// ---------------------------------------------------------------------------
static __device__ __forceinline__ void stage_kv(
    __bf16* __restrict__ Kdst, __bf16* __restrict__ Vdst,
    const __bf16* __restrict__ Ksrc,   // K rows [32][256], contiguous
    const __bf16* __restrict__ VTsrc,  // VT block base: rows [256][Sn], cols kb0..kb0+31
    int tid)
{
#pragma unroll
    for (int i = 0; i < 4; ++i) {
        const int c = tid + 256 * i;              // chunk of 8 bf16 (16 B)
        const __bf16* gk = Ksrc + c * 8;          // flat row-major copy
        __bf16*       lk = Kdst + c * 8;
        const int d    = c >> 2;                  // VT: 4 chunks per d-row
        const int col8 = (c & 3) * 8;
        const __bf16* gv = VTsrc + (size_t)d * Sn + col8;
        __bf16*       lv = Vdst + d * 32 + col8;
#if HAVE_ASYNC_LDS
        __builtin_amdgcn_global_load_async_to_lds_b128(AS1P(gk), AS3P(lk), 0, 0);
        __builtin_amdgcn_global_load_async_to_lds_b128(AS1P(gv), AS3P(lv), 0, 0);
#else
        *(bf16x8*)lk = *(const bf16x8*)gk;
        *(bf16x8*)lv = *(const bf16x8*)gv;
#endif
    }
}

__global__ __launch_bounds__(256) void flash_attn_kernel(
    const __bf16* __restrict__ Qg, const __bf16* __restrict__ Kg,
    const __bf16* __restrict__ VTg, __bf16* __restrict__ Og)
{
    __shared__ __align__(32) __bf16 Kl[2][32 * 256];   // [key][d]   2 x 16 KB
    __shared__ __align__(32) __bf16 Vl[2][256 * 32];   // [d][key]   2 x 16 KB
    __shared__ __align__(16) __bf16 Plds[8][16 * 32];  // per-wave P      8 KB

    const int tid  = threadIdx.x;
    const int wave = tid >> 5;
    const int lane = tid & 31;
    const int hsel = lane >> 4;
    const int lm   = lane & 15;

    const int b    = blockIdx.x >> 5;        // 32 query blocks per batch
    const int qblk = blockIdx.x & 31;
    const __bf16* Qb  = Qg  + (size_t)b * Sn * Dn;
    const __bf16* Kb  = Kg  + (size_t)b * Sn * Dn;
    const __bf16* VTb = VTg + (size_t)b * Dn * Sn;
    __bf16*       Ob  = Og  + (size_t)b * Sn * Dn;

    const int qrow0 = qblk * 128 + wave * 16;

    // ---- preload this wave's Q tile (16 x 256) as 8 A-fragments ----
    bf16x16 qf[8];
    {
        const __bf16* qrow = Qb + (size_t)(qrow0 + lm) * Dn;
#pragma unroll
        for (int f = 0; f < 8; ++f) {
            bf16x8 lo = *(const bf16x8*)(qrow + f * 32 + 16 * hsel);
            bf16x8 hi = *(const bf16x8*)(qrow + f * 32 + 16 * hsel + 8);
            qf[f] = joinv(lo, hi);
        }
    }

    f32x8 acc[16];
#pragma unroll
    for (int t = 0; t < 16; ++t) acc[t] = f32x8{};
    float m[8], l[8];
#pragma unroll
    for (int r = 0; r < 8; ++r) { m[r] = -1.0e30f; l[r] = 0.0f; }

    // prologue: stage key block 0 into buffer 0
    stage_kv(Kl[0], Vl[0], Kb, VTb, tid);

    for (int kb0 = 0; kb0 < Sn; kb0 += 32) {
        const int buf = (kb0 >> 5) & 1;
        const bool more = (kb0 + 32) < Sn;
        if (more)   // prefetch next key block into the other buffer
            stage_kv(Kl[buf ^ 1], Vl[buf ^ 1],
                     Kb + (size_t)(kb0 + 32) * Dn, VTb + (kb0 + 32), tid);
#if HAVE_ASYNC_LDS
        // async loads retire in order: count<=8 means the current block's
        // 8 ops (issued one iteration ago) are complete.
        if (more) __builtin_amdgcn_s_wait_asynccnt(8);
        else      __builtin_amdgcn_s_wait_asynccnt(0);
#endif
        __syncthreads();

        const __bf16* Kc = Kl[buf];
        const __bf16* Vc = Vl[buf];

        // ---- logits: two 16x16 tiles (keys kb0..+15, kb0+16..+31) ----
        f32x8 s0 = {}, s1 = {};
#pragma unroll
        for (int f = 0; f < 8; ++f) {
            bf16x16 b0 = *(const bf16x16*)(&Kc[(lm)      * 256 + f * 32 + 16 * hsel]);
            bf16x16 b1 = *(const bf16x16*)(&Kc[(16 + lm) * 256 + f * 32 + 16 * hsel]);
            s0 = WMMA_BF16(qf[f], b0, s0);
            s1 = WMMA_BF16(qf[f], b1, s1);
        }

        // ---- online softmax update ----
        float alpha[8];
#pragma unroll
        for (int r = 0; r < 8; ++r) {
            float mx = rmax16(fmaxf(s0[r], s1[r]));
            float nm = fmaxf(m[r], mx);
            alpha[r] = __expf(m[r] - nm);
            m[r] = nm;
            float p0 = __expf(s0[r] - nm);
            float p1 = __expf(s1[r] - nm);
            const int rr = r + 8 * hsel;
            Plds[wave][rr * 32 + lm]      = (__bf16)p0;
            Plds[wave][rr * 32 + 16 + lm] = (__bf16)p1;
            l[r] = l[r] * alpha[r] + rsum16(p0 + p1);
        }
#pragma unroll
        for (int t = 0; t < 16; ++t)
#pragma unroll
            for (int r = 0; r < 8; ++r) acc[t][r] *= alpha[r];

        // ---- P (C-layout) -> A-fragment via per-wave LDS scratch ----
        const __bf16* prow = &Plds[wave][lm * 32];
        bf16x8 plo = *(const bf16x8*)(prow + 16 * hsel);
        bf16x8 phi = *(const bf16x8*)(prow + 16 * hsel + 8);
        bf16x16 pf = joinv(plo, phi);

        // ---- acc += P (16x32) @ V (32 x 256), 16 N-tiles of 16 ----
#pragma unroll
        for (int t = 0; t < 16; ++t) {
            bf16x16 bv = *(const bf16x16*)(&Vc[(t * 16 + lm) * 32 + 16 * hsel]);
            acc[t] = WMMA_BF16(pf, bv, acc[t]);
        }
        __syncthreads();   // all waves done reading buf before it is overwritten
    }

    // ---- normalize and store values (bf16) ----
    float inv[8];
#pragma unroll
    for (int r = 0; r < 8; ++r) inv[r] = 1.0f / l[r];
#pragma unroll
    for (int t = 0; t < 16; ++t) {
#pragma unroll
        for (int r = 0; r < 8; ++r) {
            const int row = qrow0 + 8 * hsel + r;
            Ob[(size_t)row * Dn + t * 16 + lm] = (__bf16)(acc[t][r] * inv[r]);
        }
    }
}

// ---------------------------------------------------------------------------
// Kernel 3: out = values @ W_o^T + b_o   (f32 output)
// ---------------------------------------------------------------------------
__global__ __launch_bounds__(256) void out_proj_kernel(
    const __bf16* __restrict__ Vals, const float* __restrict__ Wo,
    const float* __restrict__ bo, float* __restrict__ out)
{
    const int wave = threadIdx.x >> 5;
    const int lane = threadIdx.x & 31;
    const int hsel = lane >> 4;
    const int lm   = lane & 15;

    const int rowTile = blockIdx.x >> 1;               // 1024 row tiles
    const int colTile = (blockIdx.x & 1) * 8 + wave;   // 16 col tiles
    const int row0 = rowTile * 16;
    const int col0 = colTile * 16;

    const __bf16* arow = Vals + (size_t)(row0 + lm) * Dn;
    const float*  brow = Wo   + (size_t)(col0 + lm) * Dn;

    f32x8 acc = {};
#pragma unroll
    for (int f = 0; f < 8; ++f) {
        bf16x8 lo = *(const bf16x8*)(arow + f * 32 + 16 * hsel);
        bf16x8 hi = *(const bf16x8*)(arow + f * 32 + 16 * hsel + 8);
        bf16x16 afrag = joinv(lo, hi);
        const float4* bp = (const float4*)(brow + f * 32 + 16 * hsel);
        bf16x16 bfrag = cvt16(bp[0], bp[1], bp[2], bp[3]);
        acc = WMMA_BF16(afrag, bfrag, acc);
    }

    const float bias = bo[col0 + lm];
#pragma unroll
    for (int r = 0; r < 8; ++r) {
        const int row = row0 + 8 * hsel + r;
        out[(size_t)row * Dn + col0 + lm] = acc[r] + bias;
    }
}

// ---------------------------------------------------------------------------
extern "C" void kernel_launch(void* const* d_in, const int* in_sizes, int n_in,
                              void* d_out, int out_size, void* d_ws, size_t ws_size,
                              hipStream_t stream) {
    const float* x    = (const float*)d_in[0];
    const float* Wqkv = (const float*)d_in[1];
    const float* bqkv = (const float*)d_in[2];
    const float* Wo   = (const float*)d_in[3];
    const float* bo   = (const float*)d_in[4];
    float* out = (float*)d_out;

    const size_t N = (size_t)Bn * Sn * Dn;   // 4,194,304 elems
    __bf16* ws  = (__bf16*)d_ws;
    __bf16* Q   = ws;                         // [b][s][d] bf16, pre-scaled
    __bf16* K   = ws + N;                     // [b][s][d] bf16
    __bf16* VT  = ws + 2 * N;                 // [b][d][s] bf16 (transposed V)
    __bf16* VAL = ws + 3 * N;                 // [b][s][d] bf16

    // 1) QKV projection: 1024 row tiles x 48 col tiles, 8 tiles/block
    qkv_proj_kernel<<<1024 * 6, 256, 0, stream>>>(x, Wqkv, bqkv, Q, K, VT);

    // 2) Flash attention: 4 batches x 32 query blocks of 128
    flash_attn_kernel<<<Bn * 32, 256, 0, stream>>>(Q, K, VT, VAL);

    // 3) Output projection: 1024 row tiles x 16 col tiles, 8 tiles/block
    out_proj_kernel<<<1024 * 2, 256, 0, stream>>>(VAL, Wo, bo, out);
}
